// LSTMModel_9268539425125
// MI455X (gfx1250) — compile-verified, matrix operand order checked by reading, not run on hardware
//
#include <hip/hip_runtime.h>

// ---------------------------------------------------------------------------
// Fused 2-layer LSTM (H=50) + FC for MI455X (gfx1250, wave32, WMMA).
//   x: [4096, 256, 7] fp32 -> out: [4096, 7] fp32
// One workgroup = 16 batch rows for all 256 timesteps; both LSTM layers and
// the FC head run entirely in LDS (~120 KB). Gate GEMMs use
// v_wmma_f32_16x16x32_bf16 (fp32 accumulate); cell state stays fp32 in regs.
//
// Gate dim padded 200 -> 256 so each of the 4 waves owns exactly 4 gate tiles
// {w, w+4, w+8, w+12}. The K-steps round-robin across the 4 independent
// accumulators, so every dependent WMMA->WMMA reuse is separated by 3
// independent WMMAs + fragment loads (fills the 5-slot bf16 hazard window
// with work instead of v_nops).
// ---------------------------------------------------------------------------

#define BATCH    4096
#define SEQ_T    256
#define IN_F     7
#define HID      50
#define NG       200    // 4*HID real gates
#define NGP      256    // padded to 16 tiles of 16
#define NT       16     // N tiles
#define NTW      4      // tiles per wave
#define K0P      64     // layer0 K: [x(7) | h0(50)] padded to 64
#define K1P      128    // layer1 K: [h0'(50) | h1(50)] padded to 128
#define ROWS     16     // batch rows per WG (WMMA M)
#define NTHREADS 128    // 4 waves

typedef __attribute__((ext_vector_type(8)))  float    v8f;
typedef __attribute__((ext_vector_type(16))) __bf16   v16bf;
typedef __attribute__((ext_vector_type(4)))  unsigned v4u;

// Assemble one 16x32-bf16 WMMA operand fragment from LDS.
// Lane L (<16): row/col = L, K = {k0..k0+7} U {k0+16..k0+23}
// Lane L (>=16): same row/col = L-16, K chunks shifted by +8.
__device__ __forceinline__ v16bf load_frag(const __bf16* rowp, int k0, int hi) {
  const int base = k0 + 8 * hi;
  union { v4u u[2]; v16bf v; } f;
  f.u[0] = *(const v4u*)(rowp + base);        // K = base .. base+7   (16B)
  f.u[1] = *(const v4u*)(rowp + base + 16);   // K = base+16 .. +23   (16B)
  return f.v;
}

__device__ __forceinline__ float fsig(float x) {
  return 1.0f / (1.0f + __expf(-x));
}
__device__ __forceinline__ float ftanh(float x) {
  x = fminf(fmaxf(x, -15.0f), 15.0f);
  float e = __expf(2.0f * x);
  return (e - 1.0f) / (e + 1.0f);
}

__global__ __launch_bounds__(NTHREADS)
void lstm2_fused_kernel(const float* __restrict__ x,
                        const float* __restrict__ w_ih0, const float* __restrict__ w_hh0,
                        const float* __restrict__ b_ih0, const float* __restrict__ b_hh0,
                        const float* __restrict__ w_ih1, const float* __restrict__ w_hh1,
                        const float* __restrict__ b_ih1, const float* __restrict__ b_hh1,
                        const float* __restrict__ w_fc,  const float* __restrict__ b_fc,
                        float* __restrict__ out) {
  // 120 KB LDS (CDNA5 allows up to 320 KB per workgroup)
  __shared__ __align__(16) __bf16 sW0[NGP * K0P];   // [w_ih0 | w_hh0 | 0]   32 KB
  __shared__ __align__(16) __bf16 sW1[NGP * K1P];   // [w_ih1 | w_hh1 | 0]   64 KB
  __shared__ __align__(16) __bf16 sA0[ROWS * K0P];  // [x_t | h0 | 0]         2 KB
  __shared__ __align__(16) __bf16 sA1[ROWS * K1P];  // [h0' | h1 | 0]         4 KB
  __shared__ __align__(16) float  sG [ROWS * NGP];  // gate pre-activations  16 KB
  __shared__ float sB0[NGP], sB1[NGP];              // folded biases          2 KB

  const int tid     = threadIdx.x;
  // wave index is uniform across the wave: pin it to an SGPR so tile control
  // stays scalar and EXEC stays all-ones through the WMMA blocks.
  const int wave    = __builtin_amdgcn_readfirstlane(tid >> 5);
  const int lane    = tid & 31;
  const int lo      = lane & 15;   // WMMA B column / A row
  const int hi      = lane >> 4;   // K-chunk select / C row-half select
  const int rowbase = blockIdx.x * ROWS;

  // ---- one-time: weights -> LDS as bf16, biases folded (b_ih + b_hh) ----
  for (int idx = tid; idx < NGP * K0P; idx += NTHREADS) {
    const int n = idx >> 6, k = idx & 63;
    float v = 0.0f;
    if (n < NG) {
      if (k < IN_F)            v = w_ih0[n * IN_F + k];
      else if (k < IN_F + HID) v = w_hh0[n * HID + (k - IN_F)];
    }
    sW0[idx] = (__bf16)v;
  }
  for (int idx = tid; idx < NGP * K1P; idx += NTHREADS) {
    const int n = idx >> 7, k = idx & 127;
    float v = 0.0f;
    if (n < NG) {
      if (k < HID)          v = w_ih1[n * HID + k];
      else if (k < 2 * HID) v = w_hh1[n * HID + (k - HID)];
    }
    sW1[idx] = (__bf16)v;
  }
  for (int idx = tid; idx < NGP; idx += NTHREADS) {
    sB0[idx] = (idx < NG) ? (b_ih0[idx] + b_hh0[idx]) : 0.0f;
    sB1[idx] = (idx < NG) ? (b_ih1[idx] + b_hh1[idx]) : 0.0f;
  }
  for (int idx = tid; idx < ROWS * K0P; idx += NTHREADS) sA0[idx] = (__bf16)0.0f;
  for (int idx = tid; idx < ROWS * K1P; idx += NTHREADS) sA1[idx] = (__bf16)0.0f;

  // fp32 cell state in registers: cell idx = tid + 128*j  (16*50 = 800 cells)
  float c0[7], c1[7];
#pragma unroll
  for (int j = 0; j < 7; ++j) { c0[j] = 0.0f; c1[j] = 0.0f; }

  __syncthreads();

  for (int t = 0; t < SEQ_T; ++t) {
    // ---- stage x_t into A0 cols [0, 7) ----
    if (tid < ROWS * IN_F) {
      const int r = tid / IN_F, k = tid - r * IN_F;
      sA0[r * K0P + k] =
          (__bf16)x[((size_t)(rowbase + r) * SEQ_T + t) * IN_F + k];
    }
    if (tid < ROWS && t + 1 < SEQ_T) {  // pull next step's x into cache
      __builtin_prefetch(&x[((size_t)(rowbase + tid) * SEQ_T + (t + 1)) * IN_F], 0, 3);
    }
    __syncthreads();  // x_t, h0(t-1), h1(t-1) visible

    // ---- layer 0 gates: [16 x 64] @ [64 x 256] -> sG ----
    {
      const __bf16* arow = sA0 + lo * K0P;
      const v16bf a0 = load_frag(arow, 0, hi);
      const v16bf a1 = load_frag(arow, 32, hi);
      int ncol[NTW]; const __bf16* wrow[NTW]; float bn[NTW];
#pragma unroll
      for (int tt = 0; tt < NTW; ++tt) {
        ncol[tt] = (wave + 4 * tt) * 16 + lo;
        wrow[tt] = sW0 + ncol[tt] * K0P;
        bn[tt]   = sB0[ncol[tt]];
      }
      v8f acc[NTW] = {};
#pragma unroll
      for (int tt = 0; tt < NTW; ++tt)   // K-step 0, 4 independent chains
        acc[tt] = __builtin_amdgcn_wmma_f32_16x16x32_bf16(false, a0, false,
                  load_frag(wrow[tt], 0, hi), (short)0, acc[tt], false, false);
#pragma unroll
      for (int tt = 0; tt < NTW; ++tt)   // K-step 1
        acc[tt] = __builtin_amdgcn_wmma_f32_16x16x32_bf16(false, a1, false,
                  load_frag(wrow[tt], 32, hi), (short)0, acc[tt], false, false);
#pragma unroll
      for (int tt = 0; tt < NTW; ++tt)   // C/D layout: M = m + 8*hi, N = ncol
#pragma unroll
        for (int m = 0; m < 8; ++m)
          sG[(8 * hi + m) * NGP + ncol[tt]] = acc[tt][m] + bn[tt];
    }
    __syncthreads();  // gates0 ready

    // ---- layer 0 cell (fp32) ----
#pragma unroll
    for (int j = 0; j < 7; ++j) {
      const int idx = tid + j * NTHREADS;
      if (idx < ROWS * HID) {
        const int row = idx / HID, hh = idx - row * HID;
        const float* g = sG + row * NGP;
        const float iv = g[hh], fv = g[HID + hh], gv = g[2 * HID + hh], ov = g[3 * HID + hh];
        const float c = fsig(fv) * c0[j] + fsig(iv) * ftanh(gv);
        c0[j] = c;
        const __bf16 hb = (__bf16)(fsig(ov) * ftanh(c));
        sA0[row * K0P + IN_F + hh] = hb;  // recurrent input for t+1
        sA1[row * K1P + hh]        = hb;  // layer-1 input for this t
      }
    }
    __syncthreads();  // h0' staged

    // ---- layer 1 gates: [16 x 128] @ [128 x 256] -> sG ----
    {
      const __bf16* arow = sA1 + lo * K1P;
      v16bf a[4];
#pragma unroll
      for (int kb = 0; kb < 4; ++kb)
        a[kb] = load_frag(arow, kb * 32, hi);
      int ncol[NTW]; const __bf16* wrow[NTW]; float bn[NTW];
#pragma unroll
      for (int tt = 0; tt < NTW; ++tt) {
        ncol[tt] = (wave + 4 * tt) * 16 + lo;
        wrow[tt] = sW1 + ncol[tt] * K1P;
        bn[tt]   = sB1[ncol[tt]];
      }
      v8f acc[NTW] = {};
#pragma unroll
      for (int kb = 0; kb < 4; ++kb)       // round-robin K-steps across the
#pragma unroll
        for (int tt = 0; tt < NTW; ++tt)   // 4 independent accumulators
          acc[tt] = __builtin_amdgcn_wmma_f32_16x16x32_bf16(false, a[kb], false,
                    load_frag(wrow[tt], kb * 32, hi), (short)0, acc[tt], false, false);
#pragma unroll
      for (int tt = 0; tt < NTW; ++tt)
#pragma unroll
        for (int m = 0; m < 8; ++m)
          sG[(8 * hi + m) * NGP + ncol[tt]] = acc[tt][m] + bn[tt];
    }
    __syncthreads();  // gates1 ready (sG reuse next iter fenced by first sync)

    // ---- layer 1 cell (fp32) ----
#pragma unroll
    for (int j = 0; j < 7; ++j) {
      const int idx = tid + j * NTHREADS;
      if (idx < ROWS * HID) {
        const int row = idx / HID, hh = idx - row * HID;
        const float* g = sG + row * NGP;
        const float iv = g[hh], fv = g[HID + hh], gv = g[2 * HID + hh], ov = g[3 * HID + hh];
        const float c = fsig(fv) * c1[j] + fsig(iv) * ftanh(gv);
        c1[j] = c;
        sA1[row * K1P + HID + hh] = (__bf16)(fsig(ov) * ftanh(c));  // h1 for t+1
      }
    }
  }
  __syncthreads();

  // ---- FC head: out[b, o] = h1[b, :] . w_fc[o, :] + b_fc[o]  (tiny) ----
  if (tid < ROWS) {
    const int row = tid;
    float hbuf[HID];
#pragma unroll
    for (int h = 0; h < HID; ++h)
      hbuf[h] = (float)sA1[row * K1P + HID + h];
#pragma unroll
    for (int o = 0; o < IN_F; ++o) {
      float s = b_fc[o];
      for (int h = 0; h < HID; ++h) s += hbuf[h] * w_fc[o * HID + h];
      out[(size_t)(rowbase + row) * IN_F + o] = s;
    }
  }
}

extern "C" void kernel_launch(void* const* d_in, const int* in_sizes, int n_in,
                              void* d_out, int out_size, void* d_ws, size_t ws_size,
                              hipStream_t stream) {
  (void)in_sizes; (void)n_in; (void)out_size; (void)d_ws; (void)ws_size;
  const float* x     = (const float*)d_in[0];
  const float* w_ih0 = (const float*)d_in[1];
  const float* w_hh0 = (const float*)d_in[2];
  const float* b_ih0 = (const float*)d_in[3];
  const float* b_hh0 = (const float*)d_in[4];
  const float* w_ih1 = (const float*)d_in[5];
  const float* w_hh1 = (const float*)d_in[6];
  const float* b_ih1 = (const float*)d_in[7];
  const float* b_hh1 = (const float*)d_in[8];
  const float* w_fc  = (const float*)d_in[9];
  const float* b_fc  = (const float*)d_in[10];
  float* out = (float*)d_out;

  dim3 grid(BATCH / ROWS);   // 256 workgroups x 16 batch rows
  dim3 block(NTHREADS);      // 4 wave32 waves
  lstm2_fused_kernel<<<grid, block, 0, stream>>>(
      x, w_ih0, w_hh0, b_ih0, b_hh0, w_ih1, w_hh1, b_ih1, b_hh1, w_fc, b_fc, out);
}
